// DeepFM_36739150250466
// MI455X (gfx1250) — compile-verified, matrix operand order checked by reading, not run on hardware
//
#include <hip/hip_runtime.h>
#include <hip/hip_bf16.h>

typedef __attribute__((ext_vector_type(16))) _Float16 v16h;
typedef __attribute__((ext_vector_type(8)))  _Float16 v8h;
typedef __attribute__((ext_vector_type(4)))  _Float16 v4h;
typedef __attribute__((ext_vector_type(8)))  float    v8f;

#define B_ROWS  16384
#define EMB     32
#define NFIELDS 12
#define DNN_IN  384
#define H0N     256
#define H1N     128

// BatchNorm inference with running mean=0, var=1: scale = 1/sqrt(1+1e-5)
#define BN_INV 0.9999950000374997f

// ---------------------------------------------------------------------------
// Kernel 0a: f32 [K,N] -> f16 [N,K] transpose (weights into B-fragment layout)
// ---------------------------------------------------------------------------
__global__ void transpose_f16(const float* __restrict__ src,
                              _Float16* __restrict__ dst, int K, int N) {
  int t = blockIdx.x * blockDim.x + threadIdx.x;
  if (t >= K * N) return;
  int k = t / N, n = t % N;
  dst[n * K + k] = (_Float16)src[t];
}

// ---------------------------------------------------------------------------
// Kernel 1: per-row feature stage. One wave32 per batch row; lane = emb dim.
// Produces dnn_in (f16, [B,384]) and partial = fm_1st + fm_2nd (f32, [B]).
// ---------------------------------------------------------------------------
struct FeatParams {
  const int*   user_id;
  const int*   target_item_id;
  const float* user_features;   // [B,15]
  const float* item_features;   // [B,38]
  const float* item_fm1[6];
  const float* user_fm1[4];
  const float* item_fm2[6];
  const float* user_fm2[4];
  const float* item_emb;        // [500000,32]
  const float* user_emb;        // [1000000,32]
  const float* fm1_dense_W;     // [6]
  const float* fm1_dense_b;     // [1]
  const float* dense_W;         // [6,384]
  const float* dense_b;         // [384]
  _Float16*    dnn_in;          // [B,384]
  float*       partial;         // [B]
};

__global__ __launch_bounds__(256) void feat_kernel(FeatParams p) {
  int wid  = (blockIdx.x * blockDim.x + threadIdx.x) >> 5;
  int lane = threadIdx.x & 31;
  if (wid >= B_ROWS) return;
  const int b = wid;
  const float* itf = p.item_features + b * 38;
  const float* usf = p.user_features + b * 15;

  float E[NFIELDS];
  float fm1 = 0.0f;

  // item cate-id fields (cols 4..7)
#pragma unroll
  for (int i = 0; i < 4; ++i) {
    int idx = (int)itf[4 + i];
    E[i] = p.item_fm2[i][idx * EMB + lane];
    fm1 += p.item_fm1[i][idx];
  }
  // item one-hot field (cols 8..17 -> table rows 1..10)
  {
    float acc = 0.0f, a1 = 0.0f;
#pragma unroll
    for (int j = 0; j < 10; ++j) {
      float c = (float)(int)itf[8 + j];
      acc += c * p.item_fm2[4][(1 + j) * EMB + lane];
      a1  += c * p.item_fm1[4][1 + j];
    }
    E[4] = acc; fm1 += a1;
  }
  // item one-hot field (cols 18..37 -> table rows 1..20)
  {
    float acc = 0.0f, a1 = 0.0f;
#pragma unroll
    for (int j = 0; j < 20; ++j) {
      float c = (float)(int)itf[18 + j];
      acc += c * p.item_fm2[5][(1 + j) * EMB + lane];
      a1  += c * p.item_fm1[5][1 + j];
    }
    E[5] = acc; fm1 += a1;
  }
  // user cate-id fields (cols 2..4)
#pragma unroll
  for (int i = 0; i < 3; ++i) {
    int idx = (int)usf[2 + i];
    E[6 + i] = p.user_fm2[i][idx * EMB + lane];
    fm1 += p.user_fm1[i][idx];
  }
  // user one-hot field (cols 5..14 -> table rows 1..10)
  {
    float acc = 0.0f, a1 = 0.0f;
#pragma unroll
    for (int j = 0; j < 10; ++j) {
      float c = (float)(int)usf[5 + j];
      acc += c * p.user_fm2[3][(1 + j) * EMB + lane];
      a1  += c * p.user_fm1[3][1 + j];
    }
    E[9] = acc; fm1 += a1;
  }
  // id embeddings
  E[10] = p.item_emb[(long)p.target_item_id[b] * EMB + lane];
  E[11] = p.user_emb[(long)p.user_id[b] * EMB + lane];

  // numeric dense features
  float df[6] = { itf[0], itf[1], itf[2], itf[3], usf[0], usf[1] };
  float fmd = p.fm1_dense_b[0];
#pragma unroll
  for (int k = 0; k < 6; ++k) fmd += df[k] * p.fm1_dense_W[k];
  fm1 += fmd;

  // FM second order: 0.5 * sum_e( s^2 - sum_f E^2 )
  float s = 0.0f, ss = 0.0f;
#pragma unroll
  for (int f = 0; f < NFIELDS; ++f) { s += E[f]; ss += E[f] * E[f]; }
  float v = s * s - ss;
#pragma unroll
  for (int off = 16; off > 0; off >>= 1) v += __shfl_xor(v, off, 32);

  // dnn_in = flatten(E) + relu(dense @ dense_W + dense_b)
#pragma unroll
  for (int f = 0; f < NFIELDS; ++f) {
    int col = f * EMB + lane;
    float d = p.dense_b[col];
#pragma unroll
    for (int k = 0; k < 6; ++k) d += df[k] * p.dense_W[k * DNN_IN + col];
    d = fmaxf(d, 0.0f);
    p.dnn_in[b * DNN_IN + col] = (_Float16)(E[f] + d);
  }
  if (lane == 0) p.partial[b] = fm1 + 0.5f * v;
}

// ---------------------------------------------------------------------------
// WMMA fragment load: 16-bit 16x32 layout (ISA 7.12.2). Caller pre-offsets
// the pointer by row*ld + (lane>>4)*8; v[0..7] from p, v[8..15] from p+16.
// ---------------------------------------------------------------------------
__device__ inline v16h load_frag16(const _Float16* p) {
  union { v16h v; v8h h[2]; } u;
  u.h[0] = *(const v8h*)(p);
  u.h[1] = *(const v8h*)(p + 16);
  return u.v;
}

// ---------------------------------------------------------------------------
// Kernel 2/3: GEMM  Out[M,N] = relu( BN( A[M,K] @ WT[N,K]^T + bias ) ), f16 out
// One wave computes a 16x64 strip: 1 A fragment reused over 4 WMMA tiles/Kstep.
// ---------------------------------------------------------------------------
__global__ __launch_bounds__(128)
void gemm_bn_relu(const _Float16* __restrict__ A, const _Float16* __restrict__ WT,
                  const float* __restrict__ bias, const float* __restrict__ gamma,
                  const float* __restrict__ beta, _Float16* __restrict__ Out,
                  int M, int N, int K) {
  int wid  = (blockIdx.x * blockDim.x + threadIdx.x) >> 5;
  int lane = threadIdx.x & 31;
  int tilesN = N >> 6;                       // 64-wide wave strips
  int totalWaves = (M >> 4) * tilesN;
  if (wid >= totalWaves) return;
  int tm = wid / tilesN, tn = wid % tilesN;

  int row  = lane & 15;
  int hi   = lane >> 4;
  int koff = hi * 8;

  const _Float16* Ar = A  + (tm * 16 + row) * K + koff;   // A rows
  const _Float16* Br = WT + (tn * 64 + row) * K + koff;   // B cols (WT rows)

  v8f acc0 = {}, acc1 = {}, acc2 = {}, acc3 = {};
  for (int k0 = 0; k0 < K; k0 += 32) {
    v16h a  = load_frag16(Ar + k0);
    v16h b0 = load_frag16(Br + k0);
    v16h b1 = load_frag16(Br + 16 * K + k0);
    v16h b2 = load_frag16(Br + 32 * K + k0);
    v16h b3 = load_frag16(Br + 48 * K + k0);
    acc0 = __builtin_amdgcn_wmma_f32_16x16x32_f16(false, a, false, b0, (short)0, acc0, false, false);
    acc1 = __builtin_amdgcn_wmma_f32_16x16x32_f16(false, a, false, b1, (short)0, acc1, false, false);
    acc2 = __builtin_amdgcn_wmma_f32_16x16x32_f16(false, a, false, b2, (short)0, acc2, false, false);
    acc3 = __builtin_amdgcn_wmma_f32_16x16x32_f16(false, a, false, b3, (short)0, acc3, false, false);
  }

  // Epilogue. C/D layout: element j of v8f -> m = tm*16 + j + hi*8, n = lane&15.
  int m0 = tm * 16 + hi * 8;
#pragma unroll
  for (int nt = 0; nt < 4; ++nt) {
    v8f acc = (nt == 0) ? acc0 : (nt == 1) ? acc1 : (nt == 2) ? acc2 : acc3;
    int n = tn * 64 + nt * 16 + row;
    float gg = gamma[n] * BN_INV;
    float bb = bias[n];
    float be = beta[n];
#pragma unroll
    for (int j = 0; j < 8; ++j) {
      float val = (acc[j] + bb) * gg + be;
      val = fmaxf(val, 0.0f);
      Out[(m0 + j) * N + n] = (_Float16)val;
    }
  }
}

// ---------------------------------------------------------------------------
// Kernel 4: out[b] = partial[b] + h1[b,:] . W2   (wave32 per row)
// ---------------------------------------------------------------------------
__global__ __launch_bounds__(256)
void final_kernel(const _Float16* __restrict__ h1, const float* __restrict__ W2,
                  const float* __restrict__ partial, float* __restrict__ out) {
  int wid  = (blockIdx.x * blockDim.x + threadIdx.x) >> 5;
  int lane = threadIdx.x & 31;
  if (wid >= B_ROWS) return;
  v4h hv = *(const v4h*)(h1 + wid * H1N + lane * 4);
  float4 w = *(const float4*)(W2 + lane * 4);
  float s = (float)hv[0] * w.x + (float)hv[1] * w.y +
            (float)hv[2] * w.z + (float)hv[3] * w.w;
#pragma unroll
  for (int off = 16; off > 0; off >>= 1) s += __shfl_xor(s, off, 32);
  if (lane == 0) out[wid] = partial[wid] + s;
}

// ---------------------------------------------------------------------------
extern "C" void kernel_launch(void* const* d_in, const int* in_sizes, int n_in,
                              void* d_out, int out_size, void* d_ws, size_t ws_size,
                              hipStream_t stream) {
  // Workspace layout (bytes):
  char* ws = (char*)d_ws;
  _Float16* dnn_in  = (_Float16*)(ws);                    // 16384*384*2 = 12,582,912
  _Float16* h0      = (_Float16*)(ws + 12582912);         // 16384*256*2 =  8,388,608
  _Float16* h1      = (_Float16*)(ws + 20971520);         // 16384*128*2 =  4,194,304
  float*    partial = (float*)   (ws + 25165824);         // 16384*4     =     65,536
  _Float16* W0T     = (_Float16*)(ws + 25231360);         // 256*384*2   =    196,608
  _Float16* W1T     = (_Float16*)(ws + 25427968);         // 128*256*2   =     65,536

  // Input order (lists flattened element-wise per setup_inputs dict order):
  //  0 user_id  1 target_item_id  2 history_item_id  3 history_len
  //  4 user_features  5 item_features
  //  6..11 item_fm1[6]  12..15 user_fm1[4]  16..21 item_fm2[6]  22..25 user_fm2[4]
  // 26 item_emb_table  27 user_emb_table
  // 28 fm1_dense_W  29 fm1_dense_b  30 dense_W  31 dense_b
  // 32 W0  33 b0  34 g0  35 be0  36 W1  37 b1  38 g1  39 be1  40 W2
  FeatParams fp;
  fp.user_id        = (const int*)d_in[0];
  fp.target_item_id = (const int*)d_in[1];
  fp.user_features  = (const float*)d_in[4];
  fp.item_features  = (const float*)d_in[5];
  for (int j = 0; j < 6; ++j) fp.item_fm1[j] = (const float*)d_in[6 + j];
  for (int j = 0; j < 4; ++j) fp.user_fm1[j] = (const float*)d_in[12 + j];
  for (int j = 0; j < 6; ++j) fp.item_fm2[j] = (const float*)d_in[16 + j];
  for (int j = 0; j < 4; ++j) fp.user_fm2[j] = (const float*)d_in[22 + j];
  fp.item_emb    = (const float*)d_in[26];
  fp.user_emb    = (const float*)d_in[27];
  fp.fm1_dense_W = (const float*)d_in[28];
  fp.fm1_dense_b = (const float*)d_in[29];
  fp.dense_W     = (const float*)d_in[30];
  fp.dense_b     = (const float*)d_in[31];
  fp.dnn_in      = dnn_in;
  fp.partial     = partial;

  const float* W0  = (const float*)d_in[32];
  const float* b0  = (const float*)d_in[33];
  const float* g0  = (const float*)d_in[34];
  const float* be0 = (const float*)d_in[35];
  const float* W1  = (const float*)d_in[36];
  const float* b1  = (const float*)d_in[37];
  const float* g1  = (const float*)d_in[38];
  const float* be1 = (const float*)d_in[39];
  const float* W2  = (const float*)d_in[40];

  // Weight transposition + f16 conversion (tiny; L2-resident afterwards)
  transpose_f16<<<(DNN_IN * H0N + 255) / 256, 256, 0, stream>>>(W0, W0T, DNN_IN, H0N);
  transpose_f16<<<(H0N * H1N + 255) / 256, 256, 0, stream>>>(W1, W1T, H0N, H1N);

  // Embedding / FM / dnn_in stage: 16384 waves, 8 per block
  feat_kernel<<<B_ROWS / 8, 256, 0, stream>>>(fp);

  // GEMM1: [16384,384] x [384,256] -> h0 ; waves = 1024*4 = 4096, 4/block
  gemm_bn_relu<<<(B_ROWS / 16) * (H0N / 64) / 4, 128, 0, stream>>>(
      dnn_in, W0T, b0, g0, be0, h0, B_ROWS, H0N, DNN_IN);

  // GEMM2: [16384,256] x [256,128] -> h1 ; waves = 1024*2 = 2048, 4/block
  gemm_bn_relu<<<(B_ROWS / 16) * (H1N / 64) / 4, 128, 0, stream>>>(
      h0, W1T, b1, g1, be1, h1, B_ROWS, H1N, H0N);

  // Final 128-dot + FM partial
  final_kernel<<<B_ROWS / 8, 256, 0, stream>>>(h1, W2, partial, (float*)d_out);
}
